// ReconChessNet_21337397527418
// MI455X (gfx1250) — compile-verified, matrix operand order checked by reading, not run on hardware
//
#include <hip/hip_runtime.h>
#include <hip/hip_bf16.h>
#include <math.h>

typedef __attribute__((ext_vector_type(16))) _Float16 v16h;
typedef __attribute__((ext_vector_type(8)))  float    v8f;

// ---------------------------------------------------------------------------
// gfx1250 async global->LDS path (ASYNCcnt). The builtin takes int4*-typed
// pointers: (v4i addrspace(1)* src, v4i addrspace(3)* dst, imm off, imm cpol).
// Guarded so the file also compiles on toolchains that lack the builtins.
// ---------------------------------------------------------------------------
#if defined(__has_builtin)
#if __has_builtin(__builtin_amdgcn_global_load_async_to_lds_b128) && \
    __has_builtin(__builtin_amdgcn_s_wait_asynccnt)
#define USE_ASYNC_LDS 1
#endif
#endif

typedef int vi4 __attribute__((__vector_size__(4 * sizeof(int))));
typedef __attribute__((address_space(1))) vi4* g_vi4p;
typedef __attribute__((address_space(3))) vi4* l_vi4p;

static __device__ __forceinline__ void async_copy_b128(const void* g, void* l) {
#ifdef USE_ASYNC_LDS
  void* gnc = (void*)g;  // builtin parameter is non-const
  __builtin_amdgcn_global_load_async_to_lds_b128((g_vi4p)(vi4*)gnc,
                                                 (l_vi4p)(vi4*)l, 0, 0);
#else
  *(uint4*)l = *(const uint4*)g;
#endif
}

template <int N>
static __device__ __forceinline__ void wait_async() {
#ifdef USE_ASYNC_LDS
  __builtin_amdgcn_s_wait_asynccnt(N);
#endif
}

// ---------------------------------------------------------------------------
// WMMA helpers (CDNA5 / gfx1250, wave32).
// A fragment: 16x32 f16, row-major source with leading dim `ld`.
//   lane L: m = L&15, k-base = 8*(L>>4); VGPR v holds K = (v>>2)*16 + kb + (v&3)*2, +1
// B fragment: loaded from a pre-transposed (N x K) buffer -> identical loader.
// C/D: lane L -> col n = L&15; VGPR r -> row m = r + 8*(L>>4).
// ---------------------------------------------------------------------------
static __device__ __forceinline__ v16h load_frag(const _Float16* src, int ld) {
  int lane = threadIdx.x & 31;
  int r = lane & 15;
  int kb = (lane >> 4) << 3;
  const _Float16* p = src + r * ld + kb;
  v16h a;
#pragma unroll
  for (int v = 0; v < 8; ++v) {
    int koff = ((v >> 2) << 4) + ((v & 3) << 1);
    a[2 * v]     = p[koff];
    a[2 * v + 1] = p[koff + 1];
  }
  return a;
}

static __device__ __forceinline__ v8f wmma_f16(v16h a, v16h b, v8f c) {
  return __builtin_amdgcn_wmma_f32_16x16x32_f16(false, a, false, b, (short)0, c,
                                                false, false);
}

static __device__ __forceinline__ float lrelu(float x) {
  return x > 0.f ? x : 0.2f * x;
}
static __device__ __forceinline__ float sigm(float x) {
  return 1.f / (1.f + expf(-x));
}

// ---------------------------------------------------------------------------
// Fused conv stack: one block per (b,t) sample, all intermediates in LDS.
// x: (4096, 13, 8, 8) NHWC (H=13, W=8, C=8).
// ---------------------------------------------------------------------------
__global__ void __launch_bounds__(128) conv_feat_kernel(
    const float* __restrict__ x,
    const float* __restrict__ W1, const float* __restrict__ b1,
    const float* __restrict__ W2, const float* __restrict__ b2,
    const float* __restrict__ W3, const float* __restrict__ b3,
    const float* __restrict__ W4, const float* __restrict__ b4,
    float* __restrict__ feat32, _Float16* __restrict__ feat16) {
  __shared__ float s_in[13 * 8 * 8];   // 832
  __shared__ float s_c1[6 * 4 * 16];   // 384
  __shared__ float s_c2[5 * 3 * 32];   // 480
  __shared__ float s_c3[2 * 1 * 64];   // 128
  const int sid = blockIdx.x;
  const int tid = threadIdx.x;
  const float* xin = x + (size_t)sid * 832;
  for (int i = tid; i < 832; i += 128) s_in[i] = xin[i];
  __syncthreads();
  // conv1: 2x2 s2, 8->16 : out (6,4,16)
  for (int o = tid; o < 384; o += 128) {
    int oc = o & 15, pos = o >> 4, ow = pos & 3, oh = pos >> 2;
    float acc = b1[oc];
#pragma unroll
    for (int kh = 0; kh < 2; ++kh)
#pragma unroll
      for (int kw = 0; kw < 2; ++kw)
        for (int ic = 0; ic < 8; ++ic)
          acc += s_in[(oh * 2 + kh) * 64 + (ow * 2 + kw) * 8 + ic] *
                 W1[((kh * 2 + kw) * 8 + ic) * 16 + oc];
    s_c1[o] = lrelu(acc);
  }
  __syncthreads();
  // conv2: 2x2 s1, 16->32 : out (5,3,32)
  for (int o = tid; o < 480; o += 128) {
    int oc = o & 31, pos = o >> 5, ow = pos % 3, oh = pos / 3;
    float acc = b2[oc];
#pragma unroll
    for (int kh = 0; kh < 2; ++kh)
#pragma unroll
      for (int kw = 0; kw < 2; ++kw)
        for (int ic = 0; ic < 16; ++ic)
          acc += s_c1[((oh + kh) * 4 + (ow + kw)) * 16 + ic] *
                 W2[((kh * 2 + kw) * 16 + ic) * 32 + oc];
    s_c2[o] = lrelu(acc);
  }
  __syncthreads();
  // conv3: 2x2 s2, 32->64 : out (2,1,64)
  for (int o = tid; o < 128; o += 128) {
    int oc = o & 63, oh = o >> 6;
    float acc = b3[oc];
#pragma unroll
    for (int kh = 0; kh < 2; ++kh)
#pragma unroll
      for (int kw = 0; kw < 2; ++kw)
        for (int ic = 0; ic < 32; ++ic)
          acc += s_c2[((oh * 2 + kh) * 3 + kw) * 32 + ic] *
                 W3[((kh * 2 + kw) * 32 + ic) * 64 + oc];
    s_c3[o] = lrelu(acc);
  }
  __syncthreads();
  // conv4: 1x1, 64->128 : out (2,1,128); channels-first flatten -> feat[oc*2+oh]
  for (int o = tid; o < 256; o += 128) {
    int oc = o & 127, oh = o >> 7;
    float acc = b4[oc];
    for (int ic = 0; ic < 64; ++ic) acc += s_c3[oh * 64 + ic] * W4[ic * 128 + oc];
    acc = lrelu(acc);
    int fi = oc * 2 + oh;
    feat32[(size_t)sid * 256 + fi] = acc;
    feat16[(size_t)sid * 256 + fi] = (_Float16)acc;
  }
}

// ---------------------------------------------------------------------------
// Transpose-convert: in (K x N) f32  ->  out (N x K) f16
// ---------------------------------------------------------------------------
__global__ void transpose_to_f16(const float* __restrict__ in,
                                 _Float16* __restrict__ out, int K, int N) {
  int idx = blockIdx.x * 256 + threadIdx.x;
  if (idx >= K * N) return;
  int k = idx / N, n = idx - k * N;
  out[(size_t)n * K + k] = (_Float16)in[idx];
}

// ---------------------------------------------------------------------------
// Generic WMMA GEMM: C(MxN) = A(MxK,f16 rowmajor) * Bt(NxK,f16 rowmajor)^T
// grid.x = M/16, grid.y = N/256, block = 128 (4 waves x 4 N-tiles each)
// ---------------------------------------------------------------------------
__global__ void __launch_bounds__(128) wmma_gemm_kernel(
    const _Float16* __restrict__ A, const _Float16* __restrict__ Bt,
    float* __restrict__ C, int N, int K) {
  const int wave = threadIdx.x >> 5;
  const int mrow = blockIdx.x * 16;
  const int ntb = blockIdx.y * 16 + wave * 4;  // in 16-col tile units
  v8f acc[4] = {};
  for (int kk = 0; kk < K; kk += 32) {
    v16h a = load_frag(A + (size_t)mrow * K + kk, K);
#pragma unroll
    for (int j = 0; j < 4; ++j) {
      const _Float16* bp = Bt + (size_t)(ntb + j) * 16 * K + kk;
      __builtin_prefetch(bp + 32, 0, 1);
      v16h b = load_frag(bp, K);
      acc[j] = wmma_f16(a, b, acc[j]);
    }
  }
  const int lane = threadIdx.x & 31;
  const int n = lane & 15, mh = (lane >> 4) << 3;
#pragma unroll
  for (int j = 0; j < 4; ++j)
#pragma unroll
    for (int r = 0; r < 8; ++r)
      C[(size_t)(mrow + mh + r) * N + (ntb + j) * 16 + n] = acc[j][r];
}

// ---------------------------------------------------------------------------
// LSTM: 4 blocks x 16 batch rows; h/c/gates resident in LDS (56 KB).
// Per step: z = h@Wr via WMMA (16x1024, K=256), gates fused in f32 VALU.
// xWk has the time-parallel input projection (no bias).
// ---------------------------------------------------------------------------
__global__ void __launch_bounds__(512) lstm_kernel(
    const float* __restrict__ xWk, const _Float16* __restrict__ WrT,
    const float* __restrict__ bias, float* __restrict__ out32,
    _Float16* __restrict__ out16) {
  __shared__ _Float16 h16[16 * 256];   // 8 KB
  __shared__ float    cst[16 * 256];   // 16 KB
  __shared__ _Float16 zbuf[16 * 1024]; // 32 KB
  const int tid = threadIdx.x;
  const int wave = tid >> 5;
  const int lane = tid & 31;
  const int b0 = blockIdx.x * 16;
  for (int i = tid; i < 16 * 256; i += 512) {
    h16[i] = (_Float16)0.f;
    cst[i] = 0.f;
  }
  __syncthreads();
  for (int t = 0; t < 64; ++t) {
    // --- z = h @ Wr : each of 16 waves computes 4 N-tiles of 16x1024 ---
    const int ntb = wave * 4;
    v8f acc[4] = {};
    for (int kk = 0; kk < 256; kk += 32) {
      v16h a = load_frag(&h16[kk], 256);
#pragma unroll
      for (int j = 0; j < 4; ++j) {
        const _Float16* bp = WrT + (size_t)(ntb + j) * 16 * 256 + kk;
        v16h b = load_frag(bp, 256);
        acc[j] = wmma_f16(a, b, acc[j]);
      }
    }
    const int n = lane & 15, mh = (lane >> 4) << 3;
#pragma unroll
    for (int j = 0; j < 4; ++j)
#pragma unroll
      for (int r = 0; r < 8; ++r)
        zbuf[(mh + r) * 1024 + (ntb + j) * 16 + n] = (_Float16)acc[j][r];
    __syncthreads();
    // --- gates: i,f,g,o at offsets 0,256,512,768 ---
    for (int u = 0; u < 8; ++u) {
      int lin = tid + u * 512;  // 0..4095 = 16 rows x 256 units
      int row = lin >> 8, j = lin & 255;
      size_t srow = (size_t)(b0 + row) * 64 + t;
      const float* xk = xWk + srow * 1024;
      float zi = xk[j]       + (float)zbuf[row * 1024 + j]       + bias[j];
      float zf = xk[j + 256] + (float)zbuf[row * 1024 + j + 256] + bias[j + 256];
      float zg = xk[j + 512] + (float)zbuf[row * 1024 + j + 512] + bias[j + 512];
      float zo = xk[j + 768] + (float)zbuf[row * 1024 + j + 768] + bias[j + 768];
      float c = cst[row * 256 + j];
      c = sigm(zf) * c + sigm(zi) * tanhf(zg);
      float h = sigm(zo) * tanhf(c);
      cst[row * 256 + j] = c;
      h16[row * 256 + j] = (_Float16)h;
      out32[srow * 256 + j] = h;
      out16[srow * 256 + j] = (_Float16)h;
    }
    __syncthreads();
  }
}

// ---------------------------------------------------------------------------
// pir head (36-way softmax) + value head; accum[0] += sum pir*log(pir)
// ---------------------------------------------------------------------------
__global__ void __launch_bounds__(64) pir_v_kernel(
    const float* __restrict__ lstm, const float* __restrict__ pirW,
    const float* __restrict__ pirb, const float* __restrict__ vW,
    const float* __restrict__ vb, const int* __restrict__ a_taken,
    float* __restrict__ p_arr, float* __restrict__ vpred,
    float* __restrict__ accum) {
  __shared__ float lg[36];
  const int row = blockIdx.x, tid = threadIdx.x;
  const float* hrow = lstm + (size_t)row * 256;
  if (tid < 36) {
    float acc = pirb[tid];
    for (int k = 0; k < 256; ++k) acc += hrow[k] * pirW[k * 36 + tid];
    lg[tid] = acc;
  }
  if (tid == 63) {
    float acc = vb[0];
    for (int k = 0; k < 256; ++k) acc += hrow[k] * vW[k];
    vpred[row] = acc;
  }
  __syncthreads();
  if (tid == 0) {
    float m = -1e30f;
    for (int i = 0; i < 36; ++i) m = fmaxf(m, lg[i]);
    float s = 0.f;
    for (int i = 0; i < 36; ++i) s += expf(lg[i] - m);
    float inv = 1.f / s, ls = logf(s), pl = 0.f;
    for (int i = 0; i < 36; ++i) {
      float p = expf(lg[i] - m) * inv;
      pl += p * ((lg[i] - m) - ls);
    }
    atomicAdd(&accum[0], pl);
    if ((row & 1) == 0) p_arr[row] = expf(lg[a_taken[row]] - m) * inv;
  }
}

// ---------------------------------------------------------------------------
// pim head, fully fused: WMMA logits (16 rows/block) -> online masked softmax
// stats (m, S=sum e^{l-m}, E=sum (l-m)e^{l-m}) -> entropy + taken-action prob,
// never materializing the 4096x4096 probability matrix. The 64 MB mask is
// streamed exactly once, double-buffered into LDS with async global->LDS DMA
// (ASYNCcnt) so its HBM latency overlaps the WMMA logits computation.
//   sum pim*log(pim) = E/S - log S ;  p(a) = e^{l_a - m} / S
// ---------------------------------------------------------------------------
__global__ void __launch_bounds__(256) pim_kernel(
    const _Float16* __restrict__ lstm16, const _Float16* __restrict__ Wt,
    const float* __restrict__ pb, const int* __restrict__ mask,
    const int* __restrict__ a_taken, float* __restrict__ p_arr,
    float* __restrict__ accum) {
  __shared__ float chunk[16 * 256];                 // 16 KB logits chunk
  __shared__ __align__(16) int smask[2][16 * 256];  // 32 KB mask double-buffer
  __shared__ float sm[16], ss[16], se[16], sla[16];
  const int tid = threadIdx.x, wave = tid >> 5, lane = tid & 31;
  const int row0 = blockIdx.x * 16;
  if (tid < 16) { sm[tid] = -1e30f; ss[tid] = 0.f; se[tid] = 0.f; sla[tid] = 0.f; }

  // Issue one 16-row x 256-col mask chunk copy (4 x b128 per thread = 16 KB).
  auto issue_mask = [&](int ch) {
    const int col0 = ch * 256;
#pragma unroll
    for (int it = 0; it < 4; ++it) {
      int slot = tid + it * 256;   // 1024 slots of 16 bytes
      int r = slot >> 6;           // row 0..15
      int c4 = (slot & 63) << 2;   // int offset, multiple of 4 (16B aligned)
      async_copy_b128(mask + (size_t)(row0 + r) * 4096 + col0 + c4,
                      &smask[ch & 1][r * 256 + c4]);
    }
  };
  issue_mask(0);

  // A (16x256) resident in VGPRs: 8 k-fragments
  v16h afrag[8];
#pragma unroll
  for (int kk = 0; kk < 8; ++kk)
    afrag[kk] = load_frag(lstm16 + (size_t)row0 * 256 + kk * 32, 256);
  __syncthreads();

  for (int ch = 0; ch < 16; ++ch) {
    const int col0 = ch * 256;
    if (ch + 1 < 16) issue_mask(ch + 1);  // overlap DMA with WMMA below
#pragma unroll
    for (int jt = 0; jt < 2; ++jt) {
      const int nt = wave * 2 + jt;        // local col-tile 0..15
      const int gcol = col0 + nt * 16;
      const _Float16* bbase = Wt + (size_t)gcol * 256;
      v8f acc = {};
#pragma unroll
      for (int kk = 0; kk < 8; ++kk) {
        __builtin_prefetch(bbase + kk * 32 + 256, 0, 1);
        v16h b = load_frag(bbase + kk * 32, 256);
        acc = wmma_f16(afrag[kk], b, acc);
      }
      const int n = lane & 15, mh = (lane >> 4) << 3;
#pragma unroll
      for (int r = 0; r < 8; ++r) chunk[(mh + r) * 256 + nt * 16 + n] = acc[r];
    }
    // Drain this chunk's mask copy (the 4 just-issued for ch+1 may remain).
    if (ch + 1 < 16) wait_async<4>(); else wait_async<0>();
    __syncthreads();
    const int* mbuf = &smask[ch & 1][0];
    // --- online stats: wave w owns rows w and w+8 ---
#pragma unroll
    for (int rr = 0; rr < 2; ++rr) {
      const int r = wave + rr * 8;
      const int grow = row0 + r;
      const int a = a_taken[grow];
      float vals[8];
      int mk[8];
      float mloc = -1e30f;
#pragma unroll
      for (int u = 0; u < 8; ++u) {
        const int cc = lane + 32 * u;
        float v = chunk[r * 256 + cc] + pb[col0 + cc];
        vals[u] = v;
        mk[u] = mbuf[r * 256 + cc];
        if (a == col0 + cc) sla[r] = v;
        if (mk[u]) mloc = fmaxf(mloc, v);
      }
      for (int off = 16; off > 0; off >>= 1)
        mloc = fmaxf(mloc, __shfl_xor(mloc, off, 32));
      float sl = 0.f, el = 0.f;
#pragma unroll
      for (int u = 0; u < 8; ++u)
        if (mk[u]) {
          float e = expf(vals[u] - mloc);
          sl += e;
          el += (vals[u] - mloc) * e;
        }
      for (int off = 16; off > 0; off >>= 1) {
        sl += __shfl_xor(sl, off, 32);
        el += __shfl_xor(el, off, 32);
      }
      if (lane == 0) {
        float m = sm[r], S = ss[r], E = se[r];
        float mn = fmaxf(m, mloc);
        float d1 = m - mn, d2 = mloc - mn;
        float e1 = expf(d1), e2 = expf(d2);
        ss[r] = S * e1 + sl * e2;
        se[r] = (E + d1 * S) * e1 + (el + d2 * sl) * e2;
        sm[r] = mn;
      }
    }
    __syncthreads();
  }
  if (tid < 16) {
    const int grow = row0 + tid;
    float S = ss[tid], E = se[tid], m = sm[tid];
    atomicAdd(&accum[1], E / S - logf(S));
    if (grow & 1) p_arr[grow] = expf(sla[tid] - m) / S;
  }
}

// ---------------------------------------------------------------------------
// GAE normalization + clipped policy-gradient loss (single block).
// accum[2] = pg_loss
// ---------------------------------------------------------------------------
__global__ void __launch_bounds__(1024) pg_kernel(
    const float* __restrict__ GAE, const float* __restrict__ lg_old,
    const float* __restrict__ p_arr, float* __restrict__ accum) {
  __shared__ float sb1[32];
  __shared__ float sb2[32];
  const int tid = threadIdx.x;
  float s = 0.f, sq = 0.f;
#pragma unroll
  for (int u = 0; u < 4; ++u) {
    float g = GAE[tid + u * 1024];
    s += g;
    sq += g * g;
  }
  for (int off = 16; off > 0; off >>= 1) {
    s += __shfl_xor(s, off, 32);
    sq += __shfl_xor(sq, off, 32);
  }
  if ((tid & 31) == 0) { sb1[tid >> 5] = s; sb2[tid >> 5] = sq; }
  __syncthreads();
  if (tid < 32) {
    s = sb1[tid];
    sq = sb2[tid];
    for (int off = 16; off > 0; off >>= 1) {
      s += __shfl_xor(s, off, 32);
      sq += __shfl_xor(sq, off, 32);
    }
    if (tid == 0) { sb1[0] = s; sb2[0] = sq; }
  }
  __syncthreads();
  const float mean = sb1[0] * (1.f / 4096.f);
  const float var = sb2[0] * (1.f / 4096.f) - mean * mean;
  const float stdv = sqrtf(fmaxf(var, 0.f)) + 1e-8f;
  __syncthreads();
  float tsum = 0.f;
#pragma unroll
  for (int u = 0; u < 4; ++u) {
    const int i = tid + u * 1024;
    float g = (GAE[i] - mean) / stdv;
    float rt = expf(logf(p_arr[i]) - lg_old[i]);
    float rtc = fminf(fmaxf(rt, 0.8f), 1.2f);
    tsum += fmaxf(-g * rt, -g * rtc);
  }
  for (int off = 16; off > 0; off >>= 1) tsum += __shfl_xor(tsum, off, 32);
  if ((tid & 31) == 0) sb1[tid >> 5] = tsum;
  __syncthreads();
  if (tid < 32) {
    tsum = sb1[tid];
    for (int off = 16; off > 0; off >>= 1) tsum += __shfl_xor(tsum, off, 32);
    if (tid == 0) accum[2] = tsum * (1.f / 4096.f);
  }
}

// ---------------------------------------------------------------------------
// Value loss over the faithful (BT x BT) broadcast: accum[3] += sum of max().
// ---------------------------------------------------------------------------
__global__ void __launch_bounds__(256) vf_kernel(
    const float* __restrict__ vpred, const float* __restrict__ ovp,
    const float* __restrict__ ret, float* __restrict__ accum) {
  __shared__ float sb[8];
  const int i = blockIdx.x, tid = threadIdx.x;
  const float vp = vpred[i];
  float s = 0.f;
  for (int j = tid; j < 4096; j += 256) {
    float o = ovp[j], r = ret[j];
    float dc = fminf(fmaxf(vp - o, -0.2f), 0.2f);
    float f1 = vp - r;      f1 *= f1;
    float f2 = (o + dc) - r; f2 *= f2;
    s += fmaxf(f1, f2);
  }
  for (int off = 16; off > 0; off >>= 1) s += __shfl_xor(s, off, 32);
  if ((tid & 31) == 0) sb[tid >> 5] = s;
  __syncthreads();
  if (tid < 8) {
    s = sb[tid];
    for (int off = 4; off > 0; off >>= 1) s += __shfl_xor(s, off, 32);
    if (tid == 0) atomicAdd(&accum[3], s);
  }
}

__global__ void finalize_kernel(const float* __restrict__ accum,
                                float* __restrict__ out) {
  const float pg = accum[2];
  const float ent = -(accum[0] + accum[1]);
  const float vf = 0.5f * accum[3] / (4096.f * 4096.f);
  out[0] = pg - ent + vf;
  out[1] = pg;
  out[2] = ent;
  out[3] = vf;
}

// ---------------------------------------------------------------------------
extern "C" void kernel_launch(void* const* d_in, const int* in_sizes, int n_in,
                              void* d_out, int out_size, void* d_ws,
                              size_t ws_size, hipStream_t stream) {
  (void)in_sizes; (void)n_in; (void)out_size; (void)ws_size;
  const float* x       = (const float*)d_in[0];
  const int*   mask    = (const int*)d_in[1];
  const float* lg_old  = (const float*)d_in[2];
  const int*   a_taken = (const int*)d_in[3];
  const float* GAE     = (const float*)d_in[4];
  const float* ovp     = (const float*)d_in[5];
  const float* ret     = (const float*)d_in[6];
  const float* W1 = (const float*)d_in[7];   const float* b1 = (const float*)d_in[8];
  const float* W2 = (const float*)d_in[9];   const float* b2 = (const float*)d_in[10];
  const float* W3 = (const float*)d_in[11];  const float* b3 = (const float*)d_in[12];
  const float* W4 = (const float*)d_in[13];  const float* b4 = (const float*)d_in[14];
  const float* lstm_k = (const float*)d_in[15];
  const float* lstm_r = (const float*)d_in[16];
  const float* lstm_b = (const float*)d_in[17];
  const float* pir_W  = (const float*)d_in[18];
  const float* pir_b  = (const float*)d_in[19];
  const float* pim_W  = (const float*)d_in[20];
  const float* pim_b  = (const float*)d_in[21];
  const float* v_W    = (const float*)d_in[22];
  const float* v_b    = (const float*)d_in[23];

  char* w = (char*)d_ws;
  float*    feat32 = (float*)w;    w += (size_t)4096 * 256 * 4;
  _Float16* feat16 = (_Float16*)w; w += (size_t)4096 * 256 * 2;
  _Float16* WkT    = (_Float16*)w; w += (size_t)1024 * 256 * 2;
  _Float16* WrT    = (_Float16*)w; w += (size_t)1024 * 256 * 2;
  _Float16* pimWT  = (_Float16*)w; w += (size_t)4096 * 256 * 2;
  float*    xWk    = (float*)w;    w += (size_t)4096 * 1024 * 4;
  float*    lstm32 = (float*)w;    w += (size_t)4096 * 256 * 4;
  _Float16* lstm16 = (_Float16*)w; w += (size_t)4096 * 256 * 2;
  float*    p_arr  = (float*)w;    w += (size_t)4096 * 4;
  float*    vpred  = (float*)w;    w += (size_t)4096 * 4;
  float*    accum  = (float*)w;    w += 64;

  (void)hipMemsetAsync(accum, 0, 64, stream);

  conv_feat_kernel<<<4096, 128, 0, stream>>>(x, W1, b1, W2, b2, W3, b3, W4, b4,
                                             feat32, feat16);
  transpose_to_f16<<<(256 * 1024 + 255) / 256, 256, 0, stream>>>(lstm_k, WkT, 256, 1024);
  transpose_to_f16<<<(256 * 1024 + 255) / 256, 256, 0, stream>>>(lstm_r, WrT, 256, 1024);
  transpose_to_f16<<<(256 * 4096 + 255) / 256, 256, 0, stream>>>(pim_W, pimWT, 256, 4096);
  // xWk = feat @ Wk  (4096x1024, K=256)
  wmma_gemm_kernel<<<dim3(256, 4), 128, 0, stream>>>(feat16, WkT, xWk, 1024, 256);
  lstm_kernel<<<4, 512, 0, stream>>>(xWk, WrT, lstm_b, lstm32, lstm16);
  pir_v_kernel<<<4096, 64, 0, stream>>>(lstm32, pir_W, pir_b, v_W, v_b, a_taken,
                                        p_arr, vpred, accum);
  pim_kernel<<<256, 256, 0, stream>>>(lstm16, pimWT, pim_b, mask, a_taken, p_arr, accum);
  pg_kernel<<<1, 1024, 0, stream>>>(GAE, lg_old, p_arr, accum);
  vf_kernel<<<4096, 256, 0, stream>>>(vpred, ovp, ret, accum);
  finalize_kernel<<<1, 1, 0, stream>>>(accum, (float*)d_out);
}
